// EEGToTextModel_76038101009039
// MI455X (gfx1250) — matrix operand order repaired and checked by reading, not compiled, not run on hardware
//
#include <hip/hip_runtime.h>
#include <hip/hip_bf16.h>
#include <math.h>

#define EPS_ 1e-5f
#define NB_  512      // batch
#define HH_  128      // hidden
#define VV_  5000     // vocab
#define VPAD 5120     // vocab padded to 64 (4x16 N-blocking)
#define TT_  20       // decode steps

#define LDSB_CELL 264  // 256 K + 8 pad (bf16) -> 528B row stride, bank-conflict free
#define LDSB_PROJ 136  // 128 K + 8 pad (bf16) -> 272B row stride

typedef __attribute__((ext_vector_type(16))) __bf16 bf16x16;
typedef __attribute__((ext_vector_type(8)))  __bf16 bf16x8;
typedef __attribute__((ext_vector_type(8)))  float  v8f;

static __device__ __forceinline__ __bf16 f2bf(float x) {
  unsigned u = __builtin_bit_cast(unsigned, x);
  unsigned r = (u + 0x7FFFu + ((u >> 16) & 1u)) >> 16;
  unsigned short s = (unsigned short)r;
  return __builtin_bit_cast(__bf16, s);
}
static __device__ __forceinline__ float sigm(float x) { return 1.0f / (1.0f + __expf(-x)); }

union FragAB { bf16x16 v; bf16x8 h[2]; };

static __device__ __forceinline__ bf16x16 load_afrag(const __bf16* arow, int k0, int half) {
  FragAB a;
  // A 16x32 bf16: elems 0-7 -> K = 8*half+0..7 ; elems 8-15 -> K = 16+8*half+0..7
  a.h[0] = *(const bf16x8*)(arow + k0 + 8 * half);
  a.h[1] = *(const bf16x8*)(arow + k0 + 16 + 8 * half);
  return a.v;
}
static __device__ __forceinline__ bf16x16 cvt_afrag_f32(const float* erow, int k0, int half) {
  FragAB a;
  const float* p0 = erow + k0 + 8 * half;
  const float* p1 = erow + k0 + 16 + 8 * half;
#pragma unroll
  for (int i = 0; i < 8; ++i) { a.v[i] = f2bf(p0[i]); a.v[8 + i] = f2bf(p1[i]); }
  return a.v;
}
static __device__ __forceinline__ bf16x16 load_bfrag(const __bf16* brow, int k0, int half) {
  FragAB b;
  // B 32x16 bf16: elems 0-7 -> K = 8*half+0..7 ; elems 8-15 -> K = 16+8*half+0..7
  b.h[0] = *(const bf16x8*)(brow + k0 + 8 * half);
  b.h[1] = *(const bf16x8*)(brow + k0 + 16 + 8 * half);
  return b.v;
}
#define WMMA_BF16(A_, B_, C_) \
  __builtin_amdgcn_wmma_f32_16x16x32_bf16(false, (A_), false, (B_), (short)0, (C_), false, false)

// ---------------------------------------------------------------------------
// Fused LSTM cell, LDS-staged weights, optional fused embedding gather.
// One wave: 16(M) x 16(gate-col) slice of ALL FOUR gate blocks (4 accumulators,
// shared A fragment), then sigmoid/tanh + c/h update entirely in registers.
//   embA != null (cell0): A cols [0,128)  = f32 emb[tok[m]] -> bf16 in regs
//                         A cols [128,256) = Afull (bf16, only h-half read)
//   embA == null (cell1): A = Afull (bf16, full 256 cols)
//   Wcat = [512 x 256] bf16 (rows: 0-127 i, 128-255 f, 256-383 g, 384-511 o)
// ---------------------------------------------------------------------------
__global__ void __launch_bounds__(128)
lstm_cell_fused(const float* __restrict__ embA, const int* __restrict__ tok,
                const __bf16* __restrict__ Afull, const __bf16* __restrict__ Wcat,
                const float* __restrict__ bc, float* __restrict__ c,
                __bf16* __restrict__ ha, __bf16* __restrict__ hb) {
  __shared__ __bf16 lbs[64 * LDSB_CELL];          // 64 weight rows x 256 K (+pad) = 33 KB
  const int lane = threadIdx.x & 31;
  const int wave = threadIdx.x >> 5;
  const int tileM = (blockIdx.x * 4 + wave) << 4; // 0..496
  const int tileN = blockIdx.y << 4;              // 0..112 (gate col block)
  const int half = lane >> 4;
  const int sub  = lane & 15;

  // stage the 4 gate-row tiles (g*128 + tileN + j, j=0..15) into LDS once
  for (int idx = threadIdx.x; idx < 64 * 32; idx += 128) {
    int row = idx >> 5, chunk = idx & 31;         // row: g*16+j ; chunk: 8-elem group
    int g = row >> 4, j = row & 15;
    const __bf16* src = Wcat + (size_t)(g * 128 + tileN + j) * 256 + chunk * 8;
    *(bf16x8*)(lbs + row * LDSB_CELL + chunk * 8) = *(const bf16x8*)src;
  }
  __syncthreads();

  const __bf16* arow = Afull + (size_t)(tileM + sub) * 256;
  const float*  erow = embA ? embA + (size_t)tok[tileM + sub] * HH_ : (const float*)nullptr;
  const __bf16* lbi = lbs + (size_t)(0 * 16 + sub) * LDSB_CELL;
  const __bf16* lbf = lbs + (size_t)(1 * 16 + sub) * LDSB_CELL;
  const __bf16* lbg = lbs + (size_t)(2 * 16 + sub) * LDSB_CELL;
  const __bf16* lbo = lbs + (size_t)(3 * 16 + sub) * LDSB_CELL;

  v8f ai = {0,0,0,0,0,0,0,0}, af = ai, ag = ai, ao = ai;
#pragma unroll
  for (int k0 = 0; k0 < 256; k0 += 32) {
    bf16x16 a = (embA && k0 < 128) ? cvt_afrag_f32(erow, k0, half)
                                   : load_afrag(arow, k0, half);
    bf16x16 b0 = load_bfrag(lbi, k0, half);
    bf16x16 b1 = load_bfrag(lbf, k0, half);
    bf16x16 b2 = load_bfrag(lbg, k0, half);
    bf16x16 b3 = load_bfrag(lbo, k0, half);
    ai = WMMA_BF16(a, b0, ai);
    af = WMMA_BF16(a, b1, af);
    ag = WMMA_BF16(a, b2, ag);
    ao = WMMA_BF16(a, b3, ao);
  }
  const int n = tileN + sub;
  const float bci = bc[n], bcf = bc[128 + n], bcg = bc[256 + n], bco = bc[384 + n];
#pragma unroll
  for (int r = 0; r < 8; ++r) {
    const int m = tileM + r + 8 * half;           // C/D layout: VGPR r -> M = r + 8*half
    float ig = sigm(ai[r] + bci);
    float fg = sigm(af[r] + bcf);
    float gg = tanhf(ag[r] + bcg);
    float og = sigm(ao[r] + bco);
    float cn = fg * c[m * HH_ + n] + ig * gg;
    c[m * HH_ + n] = cn;
    __bf16 h = f2bf(og * tanhf(cn));
    ha[m * 256 + n] = h;
    if (hb) hb[m * 256 + n] = h;
  }
}

// ---------------------------------------------------------------------------
// Projection GEMM: logits[512 x 5120] = h2[512 x 128] @ Wpb[5120 x 128]^T
// LDS-staged 64-row weight strip; one wave owns 16x64 output (4 accumulators).
// ---------------------------------------------------------------------------
__global__ void __launch_bounds__(128)
proj_gemm_fused(const __bf16* __restrict__ A /* xcat1+128, lda 256 */,
                const __bf16* __restrict__ Wpb, float* __restrict__ C) {
  __shared__ __bf16 lbs[64 * LDSB_PROJ];          // 64 rows x 128 K (+pad) = 17 KB
  const int lane = threadIdx.x & 31;
  const int wave = threadIdx.x >> 5;
  const int tileM = (blockIdx.x * 4 + wave) << 4;
  const int tileN = blockIdx.y << 6;              // 64-wide N strip

  for (int idx = threadIdx.x; idx < 64 * 16; idx += 128) {
    int row = idx >> 4, chunk = idx & 15;
    const __bf16* src = Wpb + (size_t)(tileN + row) * 128 + chunk * 8;
    *(bf16x8*)(lbs + row * LDSB_PROJ + chunk * 8) = *(const bf16x8*)src;
  }
  __syncthreads();

  const int half = lane >> 4;
  const int sub  = lane & 15;
  const __bf16* arow = A + (size_t)(tileM + sub) * 256;
  const __bf16* lb0 = lbs + (size_t)(0 * 16 + sub) * LDSB_PROJ;
  const __bf16* lb1 = lbs + (size_t)(1 * 16 + sub) * LDSB_PROJ;
  const __bf16* lb2 = lbs + (size_t)(2 * 16 + sub) * LDSB_PROJ;
  const __bf16* lb3 = lbs + (size_t)(3 * 16 + sub) * LDSB_PROJ;

  v8f a0 = {0,0,0,0,0,0,0,0}, a1 = a0, a2 = a0, a3 = a0;
#pragma unroll
  for (int k0 = 0; k0 < 128; k0 += 32) {
    bf16x16 a = load_afrag(arow, k0, half);
    bf16x16 b0 = load_bfrag(lb0, k0, half);
    bf16x16 b1 = load_bfrag(lb1, k0, half);
    bf16x16 b2 = load_bfrag(lb2, k0, half);
    bf16x16 b3 = load_bfrag(lb3, k0, half);
    a0 = WMMA_BF16(a, b0, a0);
    a1 = WMMA_BF16(a, b1, a1);
    a2 = WMMA_BF16(a, b2, a2);
    a3 = WMMA_BF16(a, b3, a3);
  }
  float* cbase = C + (size_t)(tileM + 8 * half) * VPAD + tileN + sub;
#pragma unroll
  for (int r = 0; r < 8; ++r) {
    cbase[(size_t)r * VPAD +  0] = a0[r];
    cbase[(size_t)r * VPAD + 16] = a1[r];
    cbase[(size_t)r * VPAD + 32] = a2[r];
    cbase[(size_t)r * VPAD + 48] = a3[r];
  }
}

// ---------------------------------------------------------------------------
// CNN encoder (gather-bound; plain f32 VALU, fused conv+BN+ReLU+pool)
// ---------------------------------------------------------------------------
__global__ void conv1_fused(const float* __restrict__ x, const float* __restrict__ W,
                            const float* __restrict__ bias, const float* __restrict__ g,
                            const float* __restrict__ beta, const float* __restrict__ mean,
                            const float* __restrict__ var, float* __restrict__ out) {
  int tid = blockIdx.x * blockDim.x + threadIdx.x;
  const int total = NB_ * 32 * 500;
  if (tid >= total) return;
  int lp = tid % 500;
  int co = (tid / 500) % 32;
  int b  = tid / (500 * 32);
  float scale = g[co] * rsqrtf(var[co] + EPS_);
  float shift = beta[co] - mean[co] * scale;
  const float* xb = x + (size_t)b * 19 * 3000;
  const float* wc = W + co * 19 * 7;
  float best = -INFINITY;
  for (int j = 0; j < 3; ++j) {
    int l = lp * 3 + j;
    float acc = bias[co];
    int p0 = l * 2 - 3;           // stride 2, pad 3, K=7
    for (int ci = 0; ci < 19; ++ci) {
      const float* xr = xb + ci * 3000;
      const float* wr = wc + ci * 7;
#pragma unroll
      for (int k = 0; k < 7; ++k) {
        int p = p0 + k;
        if (p >= 0 && p < 3000) acc += xr[p] * wr[k];
      }
    }
    best = fmaxf(best, fmaxf(acc * scale + shift, 0.0f));
  }
  out[tid] = best;   // (B,32,500)
}

__global__ void conv2_fused(const float* __restrict__ x, const float* __restrict__ W,
                            const float* __restrict__ bias, const float* __restrict__ g,
                            const float* __restrict__ beta, const float* __restrict__ mean,
                            const float* __restrict__ var, float* __restrict__ out) {
  int tid = blockIdx.x * blockDim.x + threadIdx.x;
  const int total = NB_ * 64 * 83;
  if (tid >= total) return;
  int lp = tid % 83;
  int co = (tid / 83) % 64;
  int b  = tid / (83 * 64);
  float scale = g[co] * rsqrtf(var[co] + EPS_);
  float shift = beta[co] - mean[co] * scale;
  const float* xb = x + (size_t)b * 32 * 500;
  const float* wc = W + co * 32 * 5;
  float best = -INFINITY;
  for (int j = 0; j < 3; ++j) {
    int l = lp * 3 + j;           // 0..248 (< 250)
    float acc = bias[co];
    int p0 = l * 2 - 2;           // stride 2, pad 2, K=5
    for (int ci = 0; ci < 32; ++ci) {
      const float* xr = xb + ci * 500;
      const float* wr = wc + ci * 5;
#pragma unroll
      for (int k = 0; k < 5; ++k) {
        int p = p0 + k;
        if (p >= 0 && p < 500) acc += xr[p] * wr[k];
      }
    }
    best = fmaxf(best, fmaxf(acc * scale + shift, 0.0f));
  }
  out[tid] = best;   // (B,64,83)
}

// conv3 + BN + ReLU + global mean -> features; write bf16 h0 for both LSTM layers
__global__ void __launch_bounds__(128)
conv3_feat(const float* __restrict__ x, const float* __restrict__ W,
           const float* __restrict__ bias, const float* __restrict__ g,
           const float* __restrict__ beta, const float* __restrict__ mean,
           const float* __restrict__ var,
           __bf16* __restrict__ xcat0, __bf16* __restrict__ xcat1) {
  __shared__ float tile[64 * 83];
  int b = blockIdx.x;
  const float* xb = x + (size_t)b * 64 * 83;
  for (int i = threadIdx.x; i < 64 * 83; i += 128) tile[i] = xb[i];
  __syncthreads();
  int co = threadIdx.x;           // 0..127
  float scale = g[co] * rsqrtf(var[co] + EPS_);
  float shift = beta[co] - mean[co] * scale;
  const float* wc = W + co * 64 * 3;
  float sum = 0.f;
  for (int l = 0; l < 42; ++l) {
    float acc = bias[co];
    int p0 = l * 2 - 1;           // stride 2, pad 1, K=3
    for (int ci = 0; ci < 64; ++ci) {
      const float* tr = tile + ci * 83;
      const float* wr = wc + ci * 3;
#pragma unroll
      for (int k = 0; k < 3; ++k) {
        int p = p0 + k;
        if (p >= 0 && p < 83) acc += tr[p] * wr[k];
      }
    }
    sum += fmaxf(acc * scale + shift, 0.0f);
  }
  __bf16 fb = f2bf(sum * (1.0f / 42.0f));
  xcat0[b * 256 + 128 + co] = fb;   // h1 init  (xcat0[:,128:] = h1)
  xcat1[b * 256 + 128 + co] = fb;   // h2 init  (xcat1 = [h1 | h2])
}

// ---------------------------------------------------------------------------
// Decoder helpers
// ---------------------------------------------------------------------------
__global__ void prep_lstm_weights(const float* __restrict__ Wih, const float* __restrict__ Whh,
                                  const float* __restrict__ bih, const float* __restrict__ bhh,
                                  __bf16* __restrict__ Wcat, float* __restrict__ bc) {
  int tid = blockIdx.x * blockDim.x + threadIdx.x;
  if (tid < 512 * 256) {
    int n = tid >> 8, k = tid & 255;
    float w = (k < 128) ? Wih[n * 128 + k] : Whh[n * 128 + (k - 128)];
    Wcat[tid] = f2bf(w);
  }
  if (tid < 512) bc[tid] = bih[tid] + bhh[tid];
}

__global__ void prep_proj_weights(const float* __restrict__ Wp, __bf16* __restrict__ Wpb) {
  int tid = blockIdx.x * blockDim.x + threadIdx.x;
  if (tid >= VPAD * HH_) return;
  int n = tid / HH_, k = tid % HH_;
  Wpb[tid] = (n < VV_) ? f2bf(Wp[n * HH_ + k]) : f2bf(0.0f);
}

__global__ void init_state(float* __restrict__ c1, float* __restrict__ c2, int* __restrict__ tok) {
  int tid = blockIdx.x * blockDim.x + threadIdx.x;
  if (tid < NB_ * HH_) { c1[tid] = 0.f; c2[tid] = 0.f; }
  if (tid < NB_) tok[tid] = 0;
}

// add bias, emit logits slab for step t, argmax -> next token
__global__ void __launch_bounds__(256)
proj_epilogue(const float* __restrict__ logits, const float* __restrict__ bp,
              float* __restrict__ out, int* __restrict__ tok, int t) {
  __shared__ float sval[256];
  __shared__ int   sidx[256];
  int b = blockIdx.x, tid = threadIdx.x;
  const float* lr = logits + (size_t)b * VPAD;
  float* orow = out + (size_t)b * TT_ * VV_ + (size_t)t * VV_;
  float best = -INFINITY; int bidx = 0x7fffffff;
  for (int v = tid; v < VV_; v += 256) {
    float val = lr[v] + bp[v];
    orow[v] = val;
    if (val > best) { best = val; bidx = v; }
  }
  sval[tid] = best; sidx[tid] = bidx;
  __syncthreads();
  for (int s = 128; s > 0; s >>= 1) {
    if (tid < s) {
      if (sval[tid + s] > sval[tid] ||
          (sval[tid + s] == sval[tid] && sidx[tid + s] < sidx[tid])) {
        sval[tid] = sval[tid + s]; sidx[tid] = sidx[tid + s];
      }
    }
    __syncthreads();
  }
  if (tid == 0) tok[b] = sidx[0];
}

// ---------------------------------------------------------------------------
extern "C" void kernel_launch(void* const* d_in, const int* in_sizes, int n_in,
                              void* d_out, int out_size, void* d_ws, size_t ws_size,
                              hipStream_t stream) {
  (void)in_sizes; (void)n_in; (void)out_size; (void)ws_size;
  const float* eeg   = (const float*)d_in[0];
  // d_in[1] = target_length (known: 20)
  const float* W1 = (const float*)d_in[2],  *b1 = (const float*)d_in[3];
  const float* g1 = (const float*)d_in[4],  *be1 = (const float*)d_in[5];
  const float* m1 = (const float*)d_in[6],  *v1 = (const float*)d_in[7];
  const float* W2 = (const float*)d_in[8],  *b2 = (const float*)d_in[9];
  const float* g2 = (const float*)d_in[10], *be2 = (const float*)d_in[11];
  const float* m2 = (const float*)d_in[12], *v2 = (const float*)d_in[13];
  const float* W3 = (const float*)d_in[14], *b3 = (const float*)d_in[15];
  const float* g3 = (const float*)d_in[16], *be3 = (const float*)d_in[17];
  const float* m3 = (const float*)d_in[18], *v3 = (const float*)d_in[19];
  const float* emb  = (const float*)d_in[20];
  const float* Wih0 = (const float*)d_in[21], *Whh0 = (const float*)d_in[22];
  const float* bih0 = (const float*)d_in[23], *bhh0 = (const float*)d_in[24];
  const float* Wih1 = (const float*)d_in[25], *Whh1 = (const float*)d_in[26];
  const float* bih1 = (const float*)d_in[27], *bhh1 = (const float*)d_in[28];
  const float* Wp = (const float*)d_in[29], *bp = (const float*)d_in[30];
  float* out = (float*)d_out;

  char* ws = (char*)d_ws;
  // persistent regions
  const size_t POOL1_B = (size_t)NB_ * 32 * 500 * 4;   // 32.77 MB
  const size_t POOL2_B = (size_t)NB_ * 64 * 83 * 4;    // 10.88 MB
  size_t off = 0;
  size_t pool1_off = off; off += (POOL1_B + 255) & ~(size_t)255;
  size_t pool2_off = off; off += (POOL2_B + 255) & ~(size_t)255;
  size_t wcat0_off = off; off += (512 * 256 * 2 + 255) & ~(size_t)255;
  size_t wcat1_off = off; off += (512 * 256 * 2 + 255) & ~(size_t)255;
  size_t wpb_off   = off; off += ((size_t)VPAD * HH_ * 2 + 255) & ~(size_t)255;
  size_t bc0_off   = off; off += (512 * 4 + 255) & ~(size_t)255;
  size_t bc1_off   = off; off += (512 * 4 + 255) & ~(size_t)255;
  // decode buffers aliased into pool1 (dead after conv2 consumes it)
  size_t log_off   = pool1_off;                                   // 10.49 MB
  size_t c1_off    = log_off + (size_t)NB_ * VPAD * 4;
  size_t c2_off    = c1_off  + (size_t)NB_ * HH_ * 4;
  size_t xc0_off   = c2_off  + (size_t)NB_ * HH_ * 4;
  size_t xc1_off   = xc0_off + (size_t)NB_ * 256 * 2;
  size_t tok_off   = xc1_off + (size_t)NB_ * 256 * 2;

  float* pool1 = (float*)(ws + pool1_off);
  float* pool2 = (float*)(ws + pool2_off);
  __bf16* Wcat0 = (__bf16*)(ws + wcat0_off);
  __bf16* Wcat1 = (__bf16*)(ws + wcat1_off);
  __bf16* Wpb   = (__bf16*)(ws + wpb_off);
  float* bc0 = (float*)(ws + bc0_off);
  float* bc1 = (float*)(ws + bc1_off);
  float* logits = (float*)(ws + log_off);
  float* c1s = (float*)(ws + c1_off);
  float* c2s = (float*)(ws + c2_off);
  __bf16* xcat0 = (__bf16*)(ws + xc0_off);   // [:,128:256) = h1 (bf16); [:,0:128) unused
  __bf16* xcat1 = (__bf16*)(ws + xc1_off);   // [h1 | h2] bf16
  int* tok = (int*)(ws + tok_off);

  // --- weight prep (bf16) ---
  prep_lstm_weights<<<(512 * 256 + 255) / 256, 256, 0, stream>>>(Wih0, Whh0, bih0, bhh0, Wcat0, bc0);
  prep_lstm_weights<<<(512 * 256 + 255) / 256, 256, 0, stream>>>(Wih1, Whh1, bih1, bhh1, Wcat1, bc1);
  prep_proj_weights<<<(VPAD * HH_ + 255) / 256, 256, 0, stream>>>(Wp, Wpb);

  // --- CNN encoder ---
  conv1_fused<<<(NB_ * 32 * 500 + 255) / 256, 256, 0, stream>>>(eeg, W1, b1, g1, be1, m1, v1, pool1);
  conv2_fused<<<(NB_ * 64 * 83 + 255) / 256, 256, 0, stream>>>(pool1, W2, b2, g2, be2, m2, v2, pool2);
  conv3_feat<<<NB_, 128, 0, stream>>>(pool2, W3, b3, g3, be3, m3, v3, xcat0, xcat1);
  init_state<<<(NB_ * HH_ + 255) / 256, 256, 0, stream>>>(c1s, c2s, tok);

  // --- autoregressive decode: 20 steps, 4 launches each ---
  for (int t = 0; t < TT_; ++t) {
    // cell0: A = [emb[tok] | h1] (embedding fused); h1 -> xcat0[:,128:] and xcat1[:,:128]
    lstm_cell_fused<<<dim3(NB_ / 64, HH_ / 16), 128, 0, stream>>>(emb, tok, xcat0, Wcat0, bc0,
                                                                  c1s, xcat0 + 128, xcat1);
    // cell1: A = [h1 | h2]; h2 -> xcat1[:,128:]
    lstm_cell_fused<<<dim3(NB_ / 64, HH_ / 16), 128, 0, stream>>>((const float*)nullptr,
                                                                  (const int*)nullptr,
                                                                  xcat1, Wcat1, bc1,
                                                                  c2s, xcat1 + 128,
                                                                  (__bf16*)nullptr);
    // projection: logits = h2 @ Wp^T  (512 x 5120, K=128), LDS-staged weights
    proj_gemm_fused<<<dim3(NB_ / 64, VPAD / 64), 128, 0, stream>>>(xcat1 + 128, Wpb, logits);
    proj_epilogue<<<NB_, 256, 0, stream>>>(logits, bp, out, tok, t);
  }
}
